// GINENetForGCOD_66486093742154
// MI455X (gfx1250) — matrix operand order repaired and checked by reading, not compiled, not run on hardware
//
#include <hip/hip_runtime.h>
#include <hip/hip_bf16.h>

#define N_NODES 100000
#define N_EDGES 1600000
#define DIN 32
#define DE 16
#define HC 64
#define NLAYERS 4
#define MH 128
#define OUTC 10
#define NGRAPH 128
#define BN_EPS 1e-5f

#define NTILES_E (N_EDGES / 16)   // 100000 edge tiles
#define NTILES_N (N_NODES / 16)   // 6250 node tiles

typedef __bf16 bf16;
typedef bf16  bfv16 __attribute__((ext_vector_type(16)));
typedef bf16  bfv8  __attribute__((ext_vector_type(8)));
typedef float fv8   __attribute__((ext_vector_type(8)));

__device__ __forceinline__ bf16 f2bf(float f) { return (bf16)f; }

__device__ __forceinline__ fv8 wmma_bf16(bfv16 a, bfv16 b, fv8 c) {
  // D = A(16x32 bf16) x B(32x16 bf16) + C(16x16 f32)
  return __builtin_amdgcn_wmma_f32_16x16x32_bf16(false, a, false, b, (short)0, c,
                                                 false, false);
}

// ---- wave32 WMMA operand layouts (CDNA5 ISA 7.12.2) ----
// A 16x32 bf16: lane L row M=L&15; element j of reg-image = K  koff + (j/8)*16 + (L/16)*8 + j%8
// B 32x16 bf16: lane L col N; element j = K  koff + (j/8)*16 + (L/16)*8 + j%8
// C/D 16x16 f32: lane L col N=L&15; reg r row M = r + 8*(L/16)

// A-layout load of (p[m][k] + q[m][k])  (xin = h + agg, GIN eps = 0), all K valid
__device__ __forceinline__ bfv16 load_A_sum(const float* p, const float* q, int ld, int koff, int lane) {
  int m = lane & 15, half = lane >> 4;
  bfv16 a;
#pragma unroll
  for (int r = 0; r < 8; ++r) {
    int k0 = koff + ((r >> 2) << 4) + (half << 3) + ((r & 3) << 1);
    a[2 * r]     = f2bf(p[m * ld + k0]     + q[m * ld + k0]);
    a[2 * r + 1] = f2bf(p[m * ld + k0 + 1] + q[m * ld + k0 + 1]);
  }
  return a;
}

// A-layout load from an LDS tile already stored as bf16 row-major [16][ld]
__device__ __forceinline__ bfv16 load_A_lds(const bf16* base, int ld, int koff, int lane) {
  int m = lane & 15, half = lane >> 4;
  bfv16 a;
#pragma unroll
  for (int r = 0; r < 8; ++r) {
    int k0 = koff + ((r >> 2) << 4) + (half << 3) + ((r & 3) << 1);
    a[2 * r]     = base[m * ld + k0];
    a[2 * r + 1] = base[m * ld + k0 + 1];
  }
  return a;
}

// ---------------- prep: pack edge features into A-register image (bf16), once ----------------
__global__ void pack_edgeA_kernel(const float* __restrict__ ea, bf16* __restrict__ eaP) {
  for (int idx = blockIdx.x * blockDim.x + threadIdx.x; idx < NTILES_E * 32;
       idx += gridDim.x * blockDim.x) {
    int lane = idx & 31, tile = idx >> 5;
    int m = lane & 15, half = lane >> 4;
    const float* row = ea + ((size_t)tile * 16 + m) * DE + half * 8;  // 8 contiguous f32
    bfv8 v;
#pragma unroll
    for (int j = 0; j < 8; ++j) v[j] = f2bf(row[j]);
    *(bfv8*)(eaP + (size_t)idx * 8) = v;
  }
}

// ---------------- prep: pack all layer weights into B-register images (bf16) ----------------
__global__ void pack_weights_kernel(const float* __restrict__ l0_ew, const float* __restrict__ l0_eb,
                                    const float* __restrict__ l0_w1, const float* __restrict__ l0_w2,
                                    const float* __restrict__ edge_w,
                                    const float* __restrict__ w1s, const float* __restrict__ w2s,
                                    bf16* __restrict__ ewP, bf16* __restrict__ w1P,
                                    bf16* __restrict__ w2P, float* __restrict__ eb0p) {
  int tid0 = blockIdx.x * blockDim.x + threadIdx.x;
  int stride = gridDim.x * blockDim.x;
  for (int i = tid0; i < NLAYERS * 2048; i += stride) {   // ewP
    int l = i >> 11, rem = i & 2047;
    int j = rem & 15, lane = (rem >> 4) & 31, nt = rem >> 9;
    int n = nt * 16 + (lane & 15);
    int k = ((j >> 3) << 4) + ((lane >> 4) << 3) + (j & 7);
    float v = 0.f;
    if (k < DE) v = (l == 0) ? ((n < DIN) ? l0_ew[n * DE + k] : 0.f)
                             : edge_w[(size_t)(l - 1) * HC * DE + n * DE + k];
    ewP[i] = f2bf(v);
  }
  for (int i = tid0; i < NLAYERS * 4096; i += stride) {   // w1P / w2P
    int l = i >> 12, rem = i & 4095;
    int j = rem & 15, lane = (rem >> 4) & 31, kc = (rem >> 9) & 1, nt = rem >> 10;
    int n = nt * 16 + (lane & 15);
    int k = kc * 32 + ((j >> 3) << 4) + ((lane >> 4) << 3) + (j & 7);
    float v1 = (l == 0) ? ((k < DIN) ? l0_w1[n * DIN + k] : 0.f)
                        : w1s[(size_t)(l - 1) * HC * HC + n * HC + k];
    float v2 = (l == 0) ? l0_w2[n * HC + k]
                        : w2s[(size_t)(l - 1) * HC * HC + n * HC + k];
    w1P[i] = f2bf(v1);
    w2P[i] = f2bf(v2);
  }
  for (int i = tid0; i < HC; i += stride) eb0p[i] = (i < DIN) ? l0_eb[i] : 0.f;
}

__global__ void pad_x_kernel(const float* __restrict__ x, float* __restrict__ hA) {
  for (size_t i = (size_t)blockIdx.x * blockDim.x + threadIdx.x;
       i < (size_t)N_NODES * HC; i += (size_t)gridDim.x * blockDim.x) {
    size_t n = i >> 6; int c = (int)(i & 63);
    hA[i] = (c < DIN) ? x[n * DIN + c] : 0.f;
  }
}

// ---------------- edge message + scatter (fused, WMMA) ----------------
// one wave per 16-edge tile: T = EA x EW^T ; m = relu(h[src]+T+eb); agg[dst] += m
__global__ void edge_msg_kernel(const float* __restrict__ h,
                                const int* __restrict__ src, const int* __restrict__ dst,
                                const bf16* __restrict__ eaP,  // packed A images
                                const bf16* __restrict__ ewP,  // packed B images (this layer)
                                const float* __restrict__ eb,  // [64]
                                float* __restrict__ agg) {
  int lane = threadIdx.x & 31;
  int wave = blockIdx.x * (blockDim.x >> 5) + (threadIdx.x >> 5);
  int e0 = wave << 4;
  int half = lane >> 4, n15 = lane & 15;

  // A: one 16B vector load; K=16..31 pad is constant zero
  bfv8 alo = *(const bfv8*)(eaP + ((size_t)wave * 32 + lane) * 8);
  bfv16 A;
#pragma unroll
  for (int j = 0; j < 8; ++j) { A[j] = alo[j]; A[8 + j] = (bf16)0.0f; }

  // 32-bit element offsets -> GVS addressing (saddr + voffset + ioffset)
  unsigned hoff[8], aoff[8];
#pragma unroll
  for (int r = 0; r < 8; ++r) {
    int e = e0 + r + (half << 3);               // rows this lane owns in D
    hoff[r] = (unsigned)src[e] * 64u + (unsigned)n15;
    aoff[r] = (unsigned)dst[e] * 64u + (unsigned)n15;
  }

  // issue all 32 gathers up front so they pipeline under the WMMAs
  float hx[4][8];
#pragma unroll
  for (int nt = 0; nt < 4; ++nt)
#pragma unroll
    for (int r = 0; r < 8; ++r) hx[nt][r] = h[hoff[r] + (unsigned)(nt * 16)];

  float bias[4];
#pragma unroll
  for (int nt = 0; nt < 4; ++nt) bias[nt] = eb[nt * 16 + n15];

#pragma unroll
  for (int nt = 0; nt < 4; ++nt) {
    bfv16 B = *(const bfv16*)(ewP + (nt * 32 + lane) * 16);
    fv8 C = {0.f, 0.f, 0.f, 0.f, 0.f, 0.f, 0.f, 0.f};
    C = wmma_bf16(A, B, C);
#pragma unroll
    for (int r = 0; r < 8; ++r) {
      float m = fmaxf(C[r] + bias[nt] + hx[nt][r], 0.f);
      unsafeAtomicAdd(&agg[aoff[r] + (unsigned)(nt * 16)], m);  // L2-resident f32 atomic
    }
  }
}

// ---------------- node MLP (2 GEMMs, bf16 LDS staging) + BN partials ----------------
__global__ void node_mlp_kernel(const float* __restrict__ h, const float* __restrict__ agg,
                                const bf16* __restrict__ w1P, const float* __restrict__ b1,
                                const bf16* __restrict__ w2P, const float* __restrict__ b2,
                                float* __restrict__ hout, float* __restrict__ bn_acc) {
  __shared__ bf16 h1s[2][16 * HC];
  int lane = threadIdx.x & 31;
  int wv = threadIdx.x >> 5;                    // 2 waves / block
  int node0 = (blockIdx.x * 2 + wv) << 4;
  int half = lane >> 4, n15 = lane & 15;
  const float* hb = h + (size_t)node0 * HC;
  const float* ab = agg + (size_t)node0 * HC;
  bfv16 A0 = load_A_sum(hb, ab, HC, 0, lane);
  bfv16 A1 = load_A_sum(hb, ab, HC, 32, lane);
#pragma unroll
  for (int nt = 0; nt < 4; ++nt) {
    bfv16 B0 = *(const bfv16*)(w1P + ((nt * 2 + 0) * 32 + lane) * 16);
    bfv16 B1 = *(const bfv16*)(w1P + ((nt * 2 + 1) * 32 + lane) * 16);
    fv8 C = {0.f, 0.f, 0.f, 0.f, 0.f, 0.f, 0.f, 0.f};
    C = wmma_bf16(A0, B0, C);
    C = wmma_bf16(A1, B1, C);
    int c = nt * 16 + n15;
    float bias = b1[c];
#pragma unroll
    for (int r = 0; r < 8; ++r)
      h1s[wv][(r + (half << 3)) * HC + c] = f2bf(fmaxf(C[r] + bias, 0.f));
  }
  __syncthreads();
  bfv16 A2 = load_A_lds(&h1s[wv][0], HC, 0, lane);
  bfv16 A3 = load_A_lds(&h1s[wv][0], HC, 32, lane);
#pragma unroll
  for (int nt = 0; nt < 4; ++nt) {
    bfv16 B0 = *(const bfv16*)(w2P + ((nt * 2 + 0) * 32 + lane) * 16);
    bfv16 B1 = *(const bfv16*)(w2P + ((nt * 2 + 1) * 32 + lane) * 16);
    fv8 C = {0.f, 0.f, 0.f, 0.f, 0.f, 0.f, 0.f, 0.f};
    C = wmma_bf16(A2, B0, C);
    C = wmma_bf16(A3, B1, C);
    int c = nt * 16 + n15;
    float bias = b2[c];
    float s1 = 0.f, s2 = 0.f;
#pragma unroll
    for (int r = 0; r < 8; ++r) {
      float v = fmaxf(C[r] + bias, 0.f);
      hout[(unsigned)(node0 + r + (half << 3)) * 64u + (unsigned)c] = v;
      s1 += v; s2 += v * v;
    }
    unsafeAtomicAdd(&bn_acc[c], s1);
    unsafeAtomicAdd(&bn_acc[HC + c], s2);
  }
}

// ---------------- BatchNorm ----------------
__global__ void bn_finalize_kernel(const float* __restrict__ bn_acc,
                                   const float* __restrict__ gamma, const float* __restrict__ beta,
                                   float* __restrict__ ss) {
  int c = threadIdx.x;
  float mu = bn_acc[c] * (1.f / N_NODES);
  float var = bn_acc[HC + c] * (1.f / N_NODES) - mu * mu;
  float sc = gamma[c] * rsqrtf(var + BN_EPS);
  ss[c] = sc;
  ss[HC + c] = beta[c] - mu * sc;
}

__global__ void bn_apply_kernel(float* __restrict__ h, const float* __restrict__ ss) {
  for (size_t i = (size_t)blockIdx.x * blockDim.x + threadIdx.x;
       i < (size_t)N_NODES * HC; i += (size_t)gridDim.x * blockDim.x) {
    int c = (int)(i & (HC - 1));
    h[i] = fmaxf(h[i] * ss[c] + ss[HC + c], 0.f);
  }
}

// ---------------- pooling + head ----------------
__global__ void pool_kernel(const float* __restrict__ h, const int* __restrict__ batch,
                            float* __restrict__ psum, float* __restrict__ pcnt) {
  for (size_t i = (size_t)blockIdx.x * blockDim.x + threadIdx.x;
       i < (size_t)N_NODES * HC; i += (size_t)gridDim.x * blockDim.x) {
    size_t n = i >> 6; int c = (int)(i & 63);
    unsigned g = (unsigned)batch[n];
    unsafeAtomicAdd(&psum[g * 64u + (unsigned)c], h[i]);
    if (c == 0) unsafeAtomicAdd(&pcnt[g], 1.f);
  }
}

__global__ void head1_kernel(const float* __restrict__ psum, const float* __restrict__ pcnt,
                             const float* __restrict__ w, const float* __restrict__ b,
                             float* __restrict__ z) {
  int idx = blockIdx.x * blockDim.x + threadIdx.x;
  if (idx >= NGRAPH * MH) return;
  int g = idx / MH, j = idx % MH;
  float inv = 1.f / fmaxf(pcnt[g], 1.f);
  float acc = b[j];
#pragma unroll 8
  for (int c = 0; c < HC; ++c) acc += psum[g * HC + c] * inv * w[j * HC + c];
  z[idx] = fmaxf(acc, 0.f);
}

__global__ void head2_kernel(const float* __restrict__ z, const float* __restrict__ w,
                             const float* __restrict__ b, float* __restrict__ out) {
  int idx = blockIdx.x * blockDim.x + threadIdx.x;
  if (idx >= NGRAPH * OUTC) return;
  int g = idx / OUTC, o = idx % OUTC;
  float acc = b[o];
#pragma unroll 8
  for (int j = 0; j < MH; ++j) acc += z[g * MH + j] * w[o * MH + j];
  out[idx] = acc;
}

extern "C" void kernel_launch(void* const* d_in, const int* in_sizes, int n_in,
                              void* d_out, int out_size, void* d_ws, size_t ws_size,
                              hipStream_t stream) {
  const float* x      = (const float*)d_in[0];
  const int*   ei     = (const int*)d_in[1];
  const float* ea     = (const float*)d_in[2];
  const int*   batch  = (const int*)d_in[3];
  const float* l0_ew  = (const float*)d_in[4];
  const float* l0_eb  = (const float*)d_in[5];
  const float* l0_w1  = (const float*)d_in[6];
  const float* l0_b1  = (const float*)d_in[7];
  const float* l0_w2  = (const float*)d_in[8];
  const float* l0_b2  = (const float*)d_in[9];
  const float* edge_w = (const float*)d_in[10];
  const float* edge_b = (const float*)d_in[11];
  const float* w1s    = (const float*)d_in[12];
  const float* b1s    = (const float*)d_in[13];
  const float* w2s    = (const float*)d_in[14];
  const float* b2s    = (const float*)d_in[15];
  const float* bn_g   = (const float*)d_in[16];
  const float* bn_b   = (const float*)d_in[17];
  const float* hw1    = (const float*)d_in[18];
  const float* hb1    = (const float*)d_in[19];
  const float* hw2    = (const float*)d_in[20];
  const float* hb2    = (const float*)d_in[21];
  const int* src = ei;
  const int* dst = ei + N_EDGES;

  // ---- workspace layout ----
  float* ws    = (float*)d_ws;
  float* hA    = ws;                                   // N*64 f32
  float* hB    = hA + (size_t)N_NODES * HC;            // N*64 f32
  float* agg   = hB + (size_t)N_NODES * HC;            // N*64 f32
  float* bnacc = agg + (size_t)N_NODES * HC;           // 128
  float* bnss  = bnacc + 2 * HC;                       // 128
  float* psum  = bnss + 2 * HC;                        // 128*64
  float* pcnt  = psum + NGRAPH * HC;                   // 128
  float* zbuf  = pcnt + NGRAPH;                        // 128*128
  float* eb0p  = zbuf + NGRAPH * MH;                   // 64
  bf16*  eaP   = (bf16*)(eb0p + HC);                   // (E/16)*32*8 bf16 = 51.2 MB
  bf16*  ewP   = eaP + (size_t)NTILES_E * 256;         // L*2048 bf16
  bf16*  w1P   = ewP + NLAYERS * 2048;                 // L*4096 bf16
  bf16*  w2P   = w1P + NLAYERS * 4096;                 // L*4096 bf16

  // ---- one-time packing (per call): edge features + weights into WMMA register images ----
  pack_edgeA_kernel<<<4096, 256, 0, stream>>>(ea, eaP);
  pack_weights_kernel<<<64, 256, 0, stream>>>(l0_ew, l0_eb, l0_w1, l0_w2,
                                              edge_w, w1s, w2s, ewP, w1P, w2P, eb0p);
  pad_x_kernel<<<2048, 256, 0, stream>>>(x, hA);

  for (int l = 0; l < NLAYERS; ++l) {
    const bf16*  ewl = ewP + (size_t)l * 2048;
    const bf16*  w1l = w1P + (size_t)l * 4096;
    const bf16*  w2l = w2P + (size_t)l * 4096;
    const float* eb  = (l == 0) ? eb0p  : edge_b + (size_t)(l - 1) * HC;
    const float* b1  = (l == 0) ? l0_b1 : b1s    + (size_t)(l - 1) * HC;
    const float* b2  = (l == 0) ? l0_b2 : b2s    + (size_t)(l - 1) * HC;
    const float* hin  = (l & 1) ? hB : hA;
    float*       hout = (l & 1) ? hA : hB;

    hipMemsetAsync(agg, 0, (size_t)N_NODES * HC * sizeof(float), stream);
    hipMemsetAsync(bnacc, 0, 2 * HC * sizeof(float), stream);
    edge_msg_kernel<<<NTILES_E / 8, 256, 0, stream>>>(hin, src, dst, eaP, ewl, eb, agg);
    node_mlp_kernel<<<NTILES_N / 2, 64, 0, stream>>>(hin, agg, w1l, b1, w2l, b2, hout, bnacc);
    bn_finalize_kernel<<<1, HC, 0, stream>>>(bnacc, bn_g + l * HC, bn_b + l * HC, bnss);
    bn_apply_kernel<<<2048, 256, 0, stream>>>(hout, bnss);
  }
  // after 4 layers (a->b->a->b->a) final activations are in hA
  hipMemsetAsync(psum, 0, (size_t)(NGRAPH * HC + NGRAPH) * sizeof(float), stream);
  pool_kernel<<<2048, 256, 0, stream>>>(hA, batch, psum, pcnt);
  head1_kernel<<<(NGRAPH * MH + 255) / 256, 256, 0, stream>>>(psum, pcnt, hw1, hb1, zbuf);
  head2_kernel<<<(NGRAPH * OUTC + 255) / 256, 256, 0, stream>>>(zbuf, hw2, hb2, (float*)d_out);
}
